// MScanBlock_69174743269732
// MI455X (gfx1250) — matrix operand order repaired and checked by reading, not compiled, not run on hardware
//
#include <hip/hip_runtime.h>
#include <stdint.h>

// ---------------------------------------------------------------------------
// Types for CDNA5 WMMA (wave32, v_wmma_f32_16x16x32_bf16)
// ---------------------------------------------------------------------------
typedef __attribute__((ext_vector_type(16))) __bf16 v16bf;
typedef __attribute__((ext_vector_type(8)))  float  v8f;

union FragBF { v16bf v; uint4 q[2]; };
union Pack8  { uint4 q; unsigned short s[8]; };

static constexpr int kP  = 16;
static constexpr int kD  = 256;
static constexpr int kE  = 384;
static constexpr int kNS = 16;
static constexpr int kC  = 40;
static constexpr int kH  = 512;
static constexpr int kW  = 512;
static constexpr int kB  = 2;
static constexpr int kJ  = 1024;                 // 32x32 patches
static constexpr int kM  = kB * kJ;              // 2048 token rows
static constexpr int kPD = kP * kP * kC;         // 10240
static constexpr int kCC = 4 * kC;               // 160 concat channels
static constexpr int kPix = kB * kH * kW;        // 524288 output pixels

__device__ __forceinline__ unsigned short f2bf(float f) {
  unsigned u = __float_as_uint(f);
  u += 0x7FFFu + ((u >> 16) & 1u);               // round-to-nearest-even
  return (unsigned short)(u >> 16);
}
__device__ __forceinline__ float softplusf(float x) {
  return (x > 20.f) ? x : log1pf(__expf(x));
}

// ---------------------------------------------------------------------------
// Scan-order permutations (row / col / diag / diag_rev), computed on device
// ---------------------------------------------------------------------------
__global__ void build_idx_k(int* __restrict__ idx, int* __restrict__ inv) {
  int t = blockIdx.x * 256 + threadIdx.x;
  if (t >= 4 * kJ) return;
  int p = t >> 10, j = t & 1023;
  int r = j >> 5, c = j & 31;
  int k;
  if (p == 0)      k = r * 32 + c;
  else if (p == 1) k = c * 32 + r;
  else {
    int s   = r + c;                                   // diagonal id 0..62
    int cum = (s <= 32) ? (s * (s + 1)) / 2
                        : 1024 - ((63 - s) * (64 - s)) / 2;
    int cnt = (s <= 31) ? (s + 1) : (63 - s);
    int off = r - ((s > 31) ? (s - 31) : 0);
    k = (p == 2) ? (cum + off) : ((1024 - cum - cnt) + off);
  }
  inv[p * kJ + j] = k;
  idx[p * kJ + k] = j;
}

// ---------------------------------------------------------------------------
// fp32 -> bf16 conversion
// ---------------------------------------------------------------------------
__global__ void f32_to_bf16_k(const float* __restrict__ src,
                              unsigned short* __restrict__ dst, int n) {
  int i = blockIdx.x * 256 + threadIdx.x;
  if (i < n) dst[i] = f2bf(src[i]);
}

// ---------------------------------------------------------------------------
// Fuse-conv weight repack: Wfk[seg][n][k] = fuse_W[n][k][ky][kx], n padded to 48
// ---------------------------------------------------------------------------
__global__ void fusew_k(const float* __restrict__ Wf, unsigned short* __restrict__ Wfk) {
  int t = blockIdx.x * 256 + threadIdx.x;
  if (t >= 9 * 48 * kCC) return;
  int seg = t / (48 * kCC), rem = t % (48 * kCC);
  int n = rem / kCC, k = rem % kCC;
  float v = (n < kC) ? Wf[((size_t)n * kCC + k) * 9 + seg] : 0.f;
  Wfk[t] = f2bf(v);
}

// ---------------------------------------------------------------------------
// Unfold feat (B,C,H,W) -> bf16 patch matrix (2048 x 10240)
// ---------------------------------------------------------------------------
__global__ void patchify_k(const float* __restrict__ feat,
                           unsigned short* __restrict__ patches) {
  int id = blockIdx.x * 256 + threadIdx.x;
  if (id >= (kM * kPD) / 4) return;
  size_t e4 = (size_t)id * 4;
  int col = (int)(e4 % kPD);
  int m   = (int)(e4 / kPD);
  int b = m >> 10, j = m & 1023, r = j >> 5, c = j & 31;
  int cch = col >> 8, rem = col & 255, i = rem >> 4, jj = rem & 15;
  const float* src = feat + (((size_t)(b * kC + cch) * kH) + (r * kP + i)) * kW + (c * kP + jj);
  float4 v = *(const float4*)src;
  uint2 o;
  o.x = (unsigned)f2bf(v.x) | ((unsigned)f2bf(v.y) << 16);
  o.y = (unsigned)f2bf(v.z) | ((unsigned)f2bf(v.w) << 16);
  *(uint2*)(patches + e4) = o;
}

// ---------------------------------------------------------------------------
// Tiled bf16 WMMA GEMM: C(MxN) = A(MxK) * W(NxK)^T + bias
// EPI 0: bias -> fp32 C ;  EPI 1: bias + residual -> fp32 C ;
// EPI 2: bias, bf16 scatter into NHWC concat buffer at channel base chanBase
// ---------------------------------------------------------------------------
template <int EPI>
__global__ __launch_bounds__(256) void gemm_bf16_k(
    const unsigned short* __restrict__ A, const unsigned short* __restrict__ W,
    const float* __restrict__ bias, const float* __restrict__ res,
    float* __restrict__ C, int M, int N, int K, int chanBase) {
  __shared__ unsigned short ldsA[128 * 40];   // 32 bf16 + 8 pad per row
  __shared__ unsigned short ldsB[128 * 40];
  const int tid  = threadIdx.x;
  const int lane = tid & 31;
  const int wave = tid >> 5;
  const int wm   = wave >> 1;                 // 0..3 (M direction)
  const int wn   = wave & 1;                  // 0..1 (N direction)
  const int half = lane >> 4;
  const int l16  = lane & 15;
  const int bm = blockIdx.y * 128;
  const int bn = blockIdx.x * 128;
  const int lrow = tid >> 1;                  // 0..127
  const int lcol = (tid & 1) << 4;            // 0 | 16

  v8f acc[2][4];
  for (int i = 0; i < 2; ++i)
    for (int j = 0; j < 4; ++j)
      for (int g = 0; g < 8; ++g) acc[i][j][g] = 0.f;

  const unsigned short* srcA = A + (size_t)(bm + lrow) * K + lcol;
  const unsigned short* srcB = W + (size_t)(bn + lrow) * K + lcol;

  for (int kb = 0; kb < K; kb += 32) {
    uint4 a0 = *(const uint4*)(srcA + kb);
    uint4 a1 = *(const uint4*)(srcA + kb + 8);
    uint4 b0 = *(const uint4*)(srcB + kb);
    uint4 b1 = *(const uint4*)(srcB + kb + 8);
    if (kb + 32 < K) {                        // global_prefetch_b8 next K tile
      __builtin_prefetch(srcA + kb + 32, 0, 0);
      __builtin_prefetch(srcB + kb + 32, 0, 0);
    }
    __syncthreads();
    *(uint4*)&ldsA[lrow * 40 + lcol]     = a0;
    *(uint4*)&ldsA[lrow * 40 + lcol + 8] = a1;
    *(uint4*)&ldsB[lrow * 40 + lcol]     = b0;
    *(uint4*)&ldsB[lrow * 40 + lcol + 8] = b1;
    __syncthreads();

    FragBF fb[4];
    #pragma unroll
    for (int nt = 0; nt < 4; ++nt) {
      int nr = (wn * 64 + nt * 16 + l16) * 40 + half * 8;
      fb[nt].q[0] = *(const uint4*)&ldsB[nr];
      fb[nt].q[1] = *(const uint4*)&ldsB[nr + 16];
    }
    #pragma unroll
    for (int mt = 0; mt < 2; ++mt) {
      FragBF fa;
      int mr = (wm * 32 + mt * 16 + l16) * 40 + half * 8;
      fa.q[0] = *(const uint4*)&ldsA[mr];
      fa.q[1] = *(const uint4*)&ldsA[mr + 16];
      #pragma unroll
      for (int nt = 0; nt < 4; ++nt) {
        acc[mt][nt] = __builtin_amdgcn_wmma_f32_16x16x32_bf16(
            false, fa.v, false, fb[nt].v, (short)0, acc[mt][nt], false, false);
      }
    }
  }

  #pragma unroll
  for (int mt = 0; mt < 2; ++mt) {
    #pragma unroll
    for (int nt = 0; nt < 4; ++nt) {
      int col = bn + wn * 64 + nt * 16 + l16;
      float bcol = bias[col];
      #pragma unroll
      for (int g = 0; g < 8; ++g) {
        int row = bm + wm * 32 + mt * 16 + half * 8 + g;   // C/D layout
        float v = acc[mt][nt][g] + bcol;
        if (EPI == 0) {
          C[(size_t)row * N + col] = v;
        } else if (EPI == 1) {
          C[(size_t)row * N + col] = v + res[(size_t)row * N + col];
        } else {
          // bf16 NHWC scatter: pixel(b, r*16+i, c*16+jj), channel chanBase+cch
          int b = row >> 10, j = row & 1023, r = j >> 5, c = j & 31;
          int cch = col >> 8, rem = col & 255, i = rem >> 4, jj = rem & 15;
          unsigned short* cbf = (unsigned short*)C;
          size_t pix = ((size_t)b << 18) + (size_t)((r * kP + i) << 9) + (c * kP + jj);
          cbf[pix * kCC + chanBase + cch] = f2bf(v);
        }
      }
    }
  }
}

// ---------------------------------------------------------------------------
// Fused 3x3 conv as WMMA implicit GEMM over NHWC bf16 concat buffer.
// Block: 128 contiguous pixels (one row segment) x 48 output ch (40 valid).
// 9 taps x (K=160 in 32-steps); LDS staged with global_load_async_to_lds_b128.
// ---------------------------------------------------------------------------
__global__ __launch_bounds__(256) void fuse_wmma_k(
    const unsigned short* __restrict__ catT,   // (B,H,W,160) bf16
    const unsigned short* __restrict__ Wfk,    // (9,48,160) bf16
    const float* __restrict__ bias,            // 40
    float* __restrict__ out) {
  __shared__ unsigned short ldsA[128 * 40];
  __shared__ unsigned short ldsB[48 * 40];
  const int tid  = threadIdx.x;
  const int lane = tid & 31;
  const int wave = tid >> 5;
  const int half = lane >> 4;
  const int l16  = lane & 15;
  const int bm   = blockIdx.x * 128;          // 128 pixels, same image row
  const int lrow = tid >> 1;
  const int lcol = (tid & 1) << 4;

  // pixel handled by this thread's A-tile row
  const int m  = bm + lrow;
  const int pb = m >> 18;
  const int py = (m >> 9) & 511;
  const int px = m & 511;

  v8f acc[3];
  for (int nt = 0; nt < 3; ++nt)
    for (int g = 0; g < 8; ++g) acc[nt][g] = 0.f;

  const unsigned aoff = (unsigned)(uintptr_t)&ldsA[lrow * 40 + lcol];
  const unsigned boff = (unsigned)(uintptr_t)&ldsB[lrow * 40 + lcol];

  for (int seg = 0; seg < 9; ++seg) {
    const int dy = seg / 3 - 1, dx = seg % 3 - 1;
    const int sy = py + dy, sx = px + dx;
    const bool valid = (sy >= 0) && (sy < kH) && (sx >= 0) && (sx < kW);
    const int cy = valid ? sy : 0, cx = valid ? sx : 0;
    const unsigned short* srcA =
        catT + (((size_t)pb << 18) + ((size_t)cy << 9) + cx) * kCC + lcol;
    const unsigned short* srcB = Wfk + ((size_t)seg * 48 + lrow) * kCC + lcol;

    for (int kb = 0; kb < kCC; kb += 32) {
      __syncthreads();                        // LDS reads of prev step done
      if (valid) {
        unsigned long long ga = (unsigned long long)(uintptr_t)(srcA + kb);
        asm volatile("global_load_async_to_lds_b128 %0, %1, off"
                     :: "v"(aoff), "v"(ga) : "memory");
        asm volatile("global_load_async_to_lds_b128 %0, %1, off offset:16"
                     :: "v"(aoff), "v"(ga) : "memory");
      } else {
        uint4 z = {0u, 0u, 0u, 0u};
        *(uint4*)&ldsA[lrow * 40 + lcol]     = z;
        *(uint4*)&ldsA[lrow * 40 + lcol + 8] = z;
      }
      if (tid < 96) {                         // 48 weight rows x 2 halves
        unsigned long long gb = (unsigned long long)(uintptr_t)(srcB + kb);
        asm volatile("global_load_async_to_lds_b128 %0, %1, off"
                     :: "v"(boff), "v"(gb) : "memory");
        asm volatile("global_load_async_to_lds_b128 %0, %1, off offset:16"
                     :: "v"(boff), "v"(gb) : "memory");
      }
      asm volatile("s_wait_asynccnt 0" ::: "memory");
      __syncthreads();

      FragBF fa;
      int mr = (wave * 16 + l16) * 40 + half * 8;
      fa.q[0] = *(const uint4*)&ldsA[mr];
      fa.q[1] = *(const uint4*)&ldsA[mr + 16];
      #pragma unroll
      for (int nt = 0; nt < 3; ++nt) {
        FragBF fb;
        int nr = (nt * 16 + l16) * 40 + half * 8;
        fb.q[0] = *(const uint4*)&ldsB[nr];
        fb.q[1] = *(const uint4*)&ldsB[nr + 16];
        acc[nt] = __builtin_amdgcn_wmma_f32_16x16x32_bf16(
            false, fa.v, false, fb.v, (short)0, acc[nt], false, false);
      }
    }
  }

  #pragma unroll
  for (int nt = 0; nt < 3; ++nt) {
    int o = nt * 16 + l16;
    if (o < kC) {
      float bo = bias[o];
      #pragma unroll
      for (int g = 0; g < 8; ++g) {
        int row = bm + wave * 16 + half * 8 + g;
        int b = row >> 18, y = (row >> 9) & 511, x = row & 511;
        out[(((size_t)(b * kC + o) * kH) + y) * kW + x] = acc[nt][g] + bo;
      }
    }
  }
}

// ---------------------------------------------------------------------------
// Sinusoidal positional embedding added to tokens
// ---------------------------------------------------------------------------
__global__ void posemb_k(float* __restrict__ tokens) {
  int id = blockIdx.x * 256 + threadIdx.x;
  if (id >= kM * kD) return;
  int m = id >> 8, d = id & 255;
  int j = m & 1023;
  float div = __expf(-9.210340371976184f * (float)((d >> 1) << 1) * (1.0f / 256.0f));
  float ang = (float)j * div;
  tokens[id] += (d & 1) ? __cosf(ang) : __sinf(ang);
}

// ---------------------------------------------------------------------------
// seq[b][k][:] = tokens[b][idx[k]][:]
// ---------------------------------------------------------------------------
__global__ void gather_k(const float* __restrict__ tokens, const int* __restrict__ idx,
                         float* __restrict__ seq) {
  int t = blockIdx.x * 256 + threadIdx.x;
  if (t >= kM * (kD / 4)) return;
  int m = t >> 6, q = t & 63;
  int b = m >> 10, k = m & 1023;
  int src = idx[k];
  ((float4*)seq)[(size_t)m * 64 + q] =
      ((const float4*)tokens)[(size_t)((b << 10) + src) * 64 + q];
}

// ---------------------------------------------------------------------------
// LayerNorm(256) -> bf16, one row per wave, wave32 shuffle reduction
// ---------------------------------------------------------------------------
__global__ __launch_bounds__(256) void layernorm_k(
    const float* __restrict__ x, const float* __restrict__ g,
    const float* __restrict__ b, unsigned short* __restrict__ out) {
  int wave = threadIdx.x >> 5, lane = threadIdx.x & 31;
  int m = blockIdx.x * 8 + wave;
  const float* row = x + (size_t)m * kD;
  float4 v0 = *(const float4*)(row + lane * 8);
  float4 v1 = *(const float4*)(row + lane * 8 + 4);
  float vv[8] = {v0.x, v0.y, v0.z, v0.w, v1.x, v1.y, v1.z, v1.w};
  float s = 0.f, ss = 0.f;
  #pragma unroll
  for (int i = 0; i < 8; ++i) { s += vv[i]; ss += vv[i] * vv[i]; }
  #pragma unroll
  for (int o = 16; o >= 1; o >>= 1) { s += __shfl_xor(s, o, 32); ss += __shfl_xor(ss, o, 32); }
  float mean = s * (1.f / 256.f);
  float var  = ss * (1.f / 256.f) - mean * mean;
  float rstd = rsqrtf(var + 1e-5f);
  Pack8 pk;
  #pragma unroll
  for (int i = 0; i < 8; ++i) {
    int d = lane * 8 + i;
    pk.s[i] = f2bf((vv[i] - mean) * rstd * g[d] + b[d]);
  }
  *(uint4*)(out + (size_t)m * kD + lane * 8) = pk.q;
}

// ---------------------------------------------------------------------------
// gate = sigmoid(silu(z))  (in place)
// ---------------------------------------------------------------------------
__global__ void gate_k(float* __restrict__ z, int n) {
  int i = blockIdx.x * 256 + threadIdx.x;
  if (i >= n) return;
  float zz = z[i];
  float si = zz / (1.f + __expf(-zz));
  z[i] = 1.f / (1.f + __expf(-si));
}

// ---------------------------------------------------------------------------
// Depthwise conv1d K=3 same-pad; reverse=1 convolves the k-reversed sequence
// ---------------------------------------------------------------------------
__global__ void conv1d_k(const float* __restrict__ x, const float* __restrict__ w,
                         const float* __restrict__ bias, float* __restrict__ xc,
                         int reverse) {
  int id = blockIdx.x * 256 + threadIdx.x;
  if (id >= kM * kE) return;
  int e = id % kE, m = id / kE;
  int b = m >> 10, k = m & 1023;
  const float* wp = w + (size_t)e * 3;
  float s = bias[e];
  #pragma unroll
  for (int t = 0; t < 3; ++t) {
    int ks = k + t - 1;
    if (ks >= 0 && ks < kJ) {
      int src = reverse ? (kJ - 1 - ks) : ks;
      s += x[((size_t)(b << 10) + src) * kE + e] * wp[t];
    }
  }
  xc[(size_t)m * kE + e] = s;
}

// ---------------------------------------------------------------------------
// Small projections: pj[m][0..63] = xc[m] @ [B;C;D;u]^T + bias (K=384)
// ---------------------------------------------------------------------------
__global__ __launch_bounds__(64) void proj4_k(
    const float* __restrict__ xc,
    const float* __restrict__ BW, const float* __restrict__ Bb,
    const float* __restrict__ CW, const float* __restrict__ Cb,
    const float* __restrict__ DW, const float* __restrict__ Db,
    const float* __restrict__ uW, const float* __restrict__ ub,
    float* __restrict__ pj) {
  __shared__ float row[kE];
  int m = blockIdx.x, t = threadIdx.x;
  for (int i = t; i < kE; i += 64) row[i] = xc[(size_t)m * kE + i];
  __syncthreads();
  int grp = t >> 4, n = t & 15;
  const float* Wsel = (grp == 0 ? BW : grp == 1 ? CW : grp == 2 ? DW : uW) + (size_t)n * kE;
  const float* bsel = (grp == 0 ? Bb : grp == 1 ? Cb : grp == 2 ? Db : ub) + n;
  float s = *bsel;
  #pragma unroll 8
  for (int i = 0; i < kE; ++i) s += row[i] * Wsel[i];
  pj[(size_t)m * 64 + t] = s;
}

// ---------------------------------------------------------------------------
// Selective-scan: log-depth Hillis-Steele over J=1024 per (b,n) chain
// ---------------------------------------------------------------------------
__global__ __launch_bounds__(1024) void ssm_scan_k(
    const float* __restrict__ pj, const float* __restrict__ A_log,
    float* __restrict__ hsC) {
  __shared__ float sa[kJ];
  __shared__ float sh[kJ];
  int b = blockIdx.x >> 4, n = blockIdx.x & 15;
  int k = threadIdx.x;
  const float* rowp = pj + ((size_t)(b << 10) + k) * 64;
  float Bt = rowp[n], Ct = rowp[16 + n], Draw = rowp[32 + n], u = rowp[48 + n];
  float A = -softplusf(A_log[n]);
  float Delta = softplusf(Draw);
  float a = __expf(Delta * A);
  float bu = (a - 1.f) / (A + 1e-6f) * Bt * u;
  sa[k] = a; sh[k] = bu;
  __syncthreads();
  for (int off = 1; off < kJ; off <<= 1) {
    float pa = 1.f, pb = 0.f;
    if (k >= off) { pa = sa[k - off]; pb = sh[k - off]; }
    float ca = sa[k], cb = sh[k];
    __syncthreads();
    sa[k] = ca * pa;
    sh[k] = ca * pb + cb;
    __syncthreads();
  }
  hsC[((size_t)(b << 10) + k) * kNS + n] = sh[k] * Ct;
}

// ---------------------------------------------------------------------------
// y[b][kk][e] (+)= hsC[b][k][:] @ roW[e][:] + rob[e];  kk = reverse? J-1-k : k
// ---------------------------------------------------------------------------
__global__ __launch_bounds__(384) void ro_accum_k(
    const float* __restrict__ hsC, const float* __restrict__ roW,
    const float* __restrict__ rob, float* __restrict__ y,
    int reverse, int accumulate) {
  __shared__ float h[kNS];
  int m = blockIdx.x, e = threadIdx.x;
  if (e < kNS) h[e] = hsC[(size_t)m * kNS + e];
  __syncthreads();
  float s = rob[e];
  #pragma unroll
  for (int n = 0; n < kNS; ++n) s += h[n] * roW[(size_t)e * kNS + n];
  int b = m >> 10, k = m & 1023;
  int kk = reverse ? (kJ - 1 - k) : k;
  float* dst = &y[((size_t)(b << 10) + kk) * kE + e];
  if (accumulate) *dst += s; else *dst = s;
}

// ---------------------------------------------------------------------------
// ybf16 = bf16(y * gate)
// ---------------------------------------------------------------------------
__global__ void gate_mul_k(const float* __restrict__ y, const float* __restrict__ g,
                           unsigned short* __restrict__ out, int n) {
  int i = blockIdx.x * 256 + threadIdx.x;
  if (i < n) out[i] = f2bf(y[i] * g[i]);
}

// ---------------------------------------------------------------------------
// tobf[b][j][:] = bf16(toperm[b][inv[j]][:])
// ---------------------------------------------------------------------------
__global__ void unperm_bf16_k(const float* __restrict__ toperm,
                              const int* __restrict__ inv,
                              unsigned short* __restrict__ tobf) {
  int t = blockIdx.x * 256 + threadIdx.x;
  if (t >= kM * (kD / 4)) return;
  int m = t >> 6, q = t & 63;
  int b = m >> 10, j = m & 1023;
  int src = inv[j];
  float4 v = ((const float4*)toperm)[(size_t)((b << 10) + src) * 64 + q];
  uint2 o;
  o.x = (unsigned)f2bf(v.x) | ((unsigned)f2bf(v.y) << 16);
  o.y = (unsigned)f2bf(v.z) | ((unsigned)f2bf(v.w) << 16);
  *(uint2*)(tobf + (size_t)m * kD + q * 4) = o;
}

// ---------------------------------------------------------------------------
// Host orchestration
// ---------------------------------------------------------------------------
extern "C" void kernel_launch(void* const* d_in, const int* in_sizes, int n_in,
                              void* d_out, int out_size, void* d_ws, size_t ws_size,
                              hipStream_t stream) {
  (void)in_sizes; (void)n_in; (void)out_size; (void)ws_size;

  const float* feat      = (const float*)d_in[0];
  const float* proj_in_W = (const float*)d_in[1];
  const float* proj_in_b = (const float*)d_in[2];
  const float* out_W     = (const float*)d_in[3];
  const float* out_b     = (const float*)d_in[4];
  const float* fuse_W    = (const float*)d_in[5];
  const float* fuse_b    = (const float*)d_in[6];
  const float* ln_g      = (const float*)d_in[7];
  const float* ln_b      = (const float*)d_in[8];
  const float* xW        = (const float*)d_in[9];
  const float* xb        = (const float*)d_in[10];
  const float* zW        = (const float*)d_in[11];
  const float* zb        = (const float*)d_in[12];
  const float* convf_W   = (const float*)d_in[13];
  const float* convf_b   = (const float*)d_in[14];
  const float* convb_W   = (const float*)d_in[15];
  const float* convb_b   = (const float*)d_in[16];
  const float* Bf_W = (const float*)d_in[17]; const float* Bf_b = (const float*)d_in[18];
  const float* Cf_W = (const float*)d_in[19]; const float* Cf_b = (const float*)d_in[20];
  const float* Df_W = (const float*)d_in[21]; const float* Df_b = (const float*)d_in[22];
  const float* Bb_W = (const float*)d_in[23]; const float* Bb_b = (const float*)d_in[24];
  const float* Cb_W = (const float*)d_in[25]; const float* Cb_b = (const float*)d_in[26];
  const float* Db_W = (const float*)d_in[27]; const float* Db_b = (const float*)d_in[28];
  const float* A_log = (const float*)d_in[29];
  const float* u_W  = (const float*)d_in[30]; const float* u_b  = (const float*)d_in[31];
  const float* ro_W = (const float*)d_in[32]; const float* ro_b = (const float*)d_in[33];
  const float* po_W = (const float*)d_in[34]; const float* po_b = (const float*)d_in[35];
  float* out = (float*)d_out;

  // ---- workspace layout -------------------------------------------------
  char* base = (char*)d_ws;
  size_t off = 0;
  auto alloc = [&](size_t bytes) -> char* {
    char* p = base + off;
    off = (off + bytes + 255) & ~(size_t)255;
    return p;
  };
  int*            idx_a   = (int*)alloc(4 * kJ * sizeof(int));
  int*            inv_a   = (int*)alloc(4 * kJ * sizeof(int));
  unsigned short* patches = (unsigned short*)alloc((size_t)kM * kPD * 2);
  unsigned short* Wproj   = (unsigned short*)alloc((size_t)kD * kPD * 2);
  unsigned short* Wout    = (unsigned short*)alloc((size_t)kPD * kD * 2);
  unsigned short* Wxb     = (unsigned short*)alloc((size_t)4 * kE * kD * 2);
  unsigned short* Wzb     = (unsigned short*)alloc((size_t)4 * kE * kD * 2);
  unsigned short* Wpob    = (unsigned short*)alloc((size_t)4 * kD * kE * 2);
  unsigned short* Wfk     = (unsigned short*)alloc((size_t)9 * 48 * kCC * 2);
  float* tokens = (float*)alloc((size_t)kM * kD * 4);
  float* seq    = (float*)alloc((size_t)kM * kD * 4);
  unsigned short* tlnb = (unsigned short*)alloc((size_t)kM * kD * 2);
  float* xbuf  = (float*)alloc((size_t)kM * kE * 4);
  float* gateb = (float*)alloc((size_t)kM * kE * 4);
  float* xcf   = (float*)alloc((size_t)kM * kE * 4);
  float* xcb   = (float*)alloc((size_t)kM * kE * 4);
  float* pjf   = (float*)alloc((size_t)kM * 64 * 4);
  float* pjb   = (float*)alloc((size_t)kM * 64 * 4);
  float* hscf  = (float*)alloc((size_t)kM * kNS * 4);
  float* hscb  = (float*)alloc((size_t)kM * kNS * 4);
  float* ybuf  = (float*)alloc((size_t)kM * kE * 4);
  unsigned short* ybf  = (unsigned short*)alloc((size_t)kM * kE * 2);
  float* toperm = (float*)alloc((size_t)kM * kD * 4);
  unsigned short* tobf = (unsigned short*)alloc((size_t)kM * kD * 2);
  unsigned short* catT = (unsigned short*)alloc((size_t)kPix * kCC * 2);  // NHWC bf16

  // ---- static prep ------------------------------------------------------
  build_idx_k<<<16, 256, 0, stream>>>(idx_a, inv_a);

  auto cvt = [&](const float* s, unsigned short* d, int n) {
    f32_to_bf16_k<<<(n + 255) / 256, 256, 0, stream>>>(s, d, n);
  };
  cvt(proj_in_W, Wproj, kD * kPD);
  cvt(out_W,     Wout,  kPD * kD);
  cvt(xW,        Wxb,   4 * kE * kD);
  cvt(zW,        Wzb,   4 * kE * kD);
  cvt(po_W,      Wpob,  4 * kD * kE);
  fusew_k<<<(9 * 48 * kCC + 255) / 256, 256, 0, stream>>>(fuse_W, Wfk);

  patchify_k<<<(kM * kPD / 4 + 255) / 256, 256, 0, stream>>>(feat, patches);

  // tokens = patches @ proj_in_W^T + b  (M=2048, N=256, K=10240) + pos emb
  gemm_bf16_k<0><<<dim3(kD / 128, kM / 128), 256, 0, stream>>>(
      patches, Wproj, proj_in_b, nullptr, tokens, kM, kD, kPD, 0);
  posemb_k<<<(kM * kD + 255) / 256, 256, 0, stream>>>(tokens);

  // ---- per scan path ----------------------------------------------------
  for (int p = 0; p < 4; ++p) {
    const int* idx_p = idx_a + p * kJ;
    const int* inv_p = inv_a + p * kJ;

    gather_k<<<(kM * kD / 4 + 255) / 256, 256, 0, stream>>>(tokens, idx_p, seq);
    layernorm_k<<<kM / 8, 256, 0, stream>>>(seq, ln_g + p * kD, ln_b + p * kD, tlnb);

    // x = LN @ xW^T + xb ; z -> gate   (M=2048, N=384, K=256)
    gemm_bf16_k<0><<<dim3(kE / 128, kM / 128), 256, 0, stream>>>(
        tlnb, Wxb + (size_t)p * kE * kD, xb + p * kE, nullptr, xbuf, kM, kE, kD, 0);
    gemm_bf16_k<0><<<dim3(kE / 128, kM / 128), 256, 0, stream>>>(
        tlnb, Wzb + (size_t)p * kE * kD, zb + p * kE, nullptr, gateb, kM, kE, kD, 0);
    gate_k<<<(kM * kE + 255) / 256, 256, 0, stream>>>(gateb, kM * kE);

    // depthwise conv (forward / reversed)
    conv1d_k<<<(kM * kE + 255) / 256, 256, 0, stream>>>(
        xbuf, convf_W + (size_t)p * kE * 3, convf_b + p * kE, xcf, 0);
    conv1d_k<<<(kM * kE + 255) / 256, 256, 0, stream>>>(
        xbuf, convb_W + (size_t)p * kE * 3, convb_b + p * kE, xcb, 1);

    // B,C,Delta,u projections
    proj4_k<<<kM, 64, 0, stream>>>(
        xcf, Bf_W + (size_t)p * kNS * kE, Bf_b + p * kNS,
        Cf_W + (size_t)p * kNS * kE, Cf_b + p * kNS,
        Df_W + (size_t)p * kNS * kE, Df_b + p * kNS,
        u_W + (size_t)p * kNS * kE,  u_b + p * kNS, pjf);
    proj4_k<<<kM, 64, 0, stream>>>(
        xcb, Bb_W + (size_t)p * kNS * kE, Bb_b + p * kNS,
        Cb_W + (size_t)p * kNS * kE, Cb_b + p * kNS,
        Db_W + (size_t)p * kNS * kE, Db_b + p * kNS,
        u_W + (size_t)p * kNS * kE,  u_b + p * kNS, pjb);

    // selective scans (32 chains each, J=1024 log-depth)
    ssm_scan_k<<<kB * kNS, kJ, 0, stream>>>(pjf, A_log + p * kNS, hscf);
    ssm_scan_k<<<kB * kNS, kJ, 0, stream>>>(pjb, A_log + p * kNS, hscb);

    // y = yf + reverse(yb), then gate
    ro_accum_k<<<kM, kE, 0, stream>>>(hscf, ro_W + (size_t)p * kE * kNS,
                                      ro_b + p * kE, ybuf, 0, 0);
    ro_accum_k<<<kM, kE, 0, stream>>>(hscb, ro_W + (size_t)p * kE * kNS,
                                      ro_b + p * kE, ybuf, 1, 1);
    gate_mul_k<<<(kM * kE + 255) / 256, 256, 0, stream>>>(ybuf, gateb, ybf, kM * kE);

    // to = seq + y @ po_W^T + po_b  (M=2048, N=256, K=384), then un-permute
    gemm_bf16_k<1><<<dim3(kD / 128, kM / 128), 256, 0, stream>>>(
        ybf, Wpob + (size_t)p * kD * kE, po_b + p * kD, seq, toperm, kM, kD, kE, 0);
    unperm_bf16_k<<<(kM * kD / 4 + 255) / 256, 256, 0, stream>>>(toperm, inv_p, tobf);

    // pf = to @ out_W^T + out_b, scattered as bf16 into NHWC concat buffer
    gemm_bf16_k<2><<<dim3(kPD / 128, kM / 128), 256, 0, stream>>>(
        tobf, Wout, out_b, nullptr, (float*)catT, kM, kPD, kD, p * kC);
  }

  // ---- fused 3x3 conv (WMMA implicit GEMM, async LDS staging) -----------
  fuse_wmma_k<<<kPix / 128, 256, 0, stream>>>(catT, Wfk, fuse_b, out);
}
